// EvolvingAttentionModule_47553877901326
// MI455X (gfx1250) — matrix operand order repaired and checked by reading, not compiled, not run on hardware
//
#include <hip/hip_runtime.h>
#include <hip/hip_bf16.h>

// ---------------------------------------------------------------------------
// Problem sizes (fixed by the reference)
//   x: (16, 256, 3, 30, 64) f32 ; pool -> g (16,256)
//   mask = g @ Wc^T + conv_b              (Wc[o,i] = conv_w[o,i,1])
//   gi   = mask @ w_ih^T + b_ih           (16 x 768)
//   256 GRU steps: gh = h @ w_hh^T (+b_hh), gates, h update
//   out: (16, 256 steps, 256) f32
// ---------------------------------------------------------------------------

typedef __attribute__((ext_vector_type(16))) __bf16        v16bf;
typedef __attribute__((ext_vector_type(8)))  float         v8f;
typedef __attribute__((ext_vector_type(8)))  unsigned int  v8u;

#define POOL_N   5760      // 3*30*64
#define BT       4096      // 16*256
#define T_DIM    256
#define G3       768       // 3*T

__device__ __forceinline__ unsigned short f2bf(float f) {
    unsigned int u = __float_as_uint(f);
    // round-to-nearest-even bf16
    u = (u + 0x7FFFu + ((u >> 16) & 1u)) >> 16;
    return (unsigned short)u;
}

// A fragment (16x32 bf16) built from a row-major f32 matrix with on-the-fly
// conversion (used only in the one-time prologue GEMMs).
// lane L: M = L%16, half = L/16.
// VGPR v: K = k0 + 16*(v>>2) + 8*half + 2*(v&3) .. +1 (packed lo|hi)
__device__ __forceinline__ v16bf make_a_frag_f32(const float* S, int ld,
                                                 int lane, int k0) {
    int m = lane & 15, half = lane >> 4;
    v8u u;
#pragma unroll
    for (int v = 0; v < 8; ++v) {
        int k = k0 + ((v >> 2) << 4) + (half << 3) + ((v & 3) << 1);
        unsigned lo = f2bf(S[m * ld + k]);
        unsigned hi = f2bf(S[m * ld + k + 1]);
        u[v] = lo | (hi << 16);
    }
    return __builtin_bit_cast(v16bf, u);
}

// A fragment from LDS holding packed bf16 pairs (row-major, 128 u32 per
// 256-wide row). VGPRs 0-3 and 4-7 each cover 4 consecutive u32s, 16B
// aligned -> exactly two ds_load_b128 per fragment, zero conversion VALU.
__device__ __forceinline__ v16bf load_a_packed(const unsigned int* Hu,
                                               int lane, int k0) {
    int m = lane & 15, half = lane >> 4;
    const uint4* p = (const uint4*)(Hu + m * 128 + (k0 >> 1) + half * 4);
    uint4 lo = p[0];   // u32 offsets 0..3   (K group 0)
    uint4 hi = p[2];   // u32 offsets 8..11  (K group 1)
    v8u u;
    u[0] = lo.x; u[1] = lo.y; u[2] = lo.z; u[3] = lo.w;
    u[4] = hi.x; u[5] = hi.y; u[6] = hi.z; u[7] = hi.w;
    return __builtin_bit_cast(v16bf, u);
}

// B fragment (32x16 bf16) from row-major bf16 weights W[n, k] (256 cols ->
// 128 u32 per row). lane L: N = n0 + L%16; VGPR v: K = k0 + 16*(v>>2) + 8*(L/16) + 2*(v&3).
// two groups of 4 consecutive u32s, 16B aligned -> two global_load_b128.
__device__ __forceinline__ v16bf load_b_frag(const unsigned int* Wu,
                                             int lane, int n0, int k0) {
    int n = n0 + (lane & 15), half = lane >> 4;
    const uint4* p = (const uint4*)(Wu + (size_t)n * 128 + (k0 >> 1) + half * 4);
    uint4 lo = p[0];
    uint4 hi = p[2];
    v8u u;
    u[0] = lo.x; u[1] = lo.y; u[2] = lo.z; u[3] = lo.w;
    u[4] = hi.x; u[5] = hi.y; u[6] = hi.z; u[7] = hi.w;
    return __builtin_bit_cast(v16bf, u);
}

__device__ __forceinline__ v8f wmma_bf16(v16bf a, v16bf b, v8f c) {
    return __builtin_amdgcn_wmma_f32_16x16x32_bf16(
        /*neg_a=*/false, a, /*neg_b=*/false, b,
        /*c_mod=*/(short)0, c, /*reuse_a=*/false, /*reuse_b=*/false);
}

__device__ __forceinline__ float sigm(float x) {
    return 1.0f / (1.0f + __expf(-x));
}

// ---------------------------------------------------------------------------
// Kernel 1: global average pool over (D,H,W). One block per (b,t).
// 94.4 MB streamed; float4 coalesced; HBM-bound (~4 us at 23.3 TB/s).
// ---------------------------------------------------------------------------
__global__ void pool_kernel(const float* __restrict__ x, float* __restrict__ g) {
    __shared__ float red[256];
    int bt = blockIdx.x;
    const float4* p = (const float4*)(x + (size_t)bt * POOL_N);  // 1440 float4
    float s = 0.0f;
    for (int i = threadIdx.x; i < POOL_N / 4; i += 256) {
        float4 v = p[i];
        s += v.x + v.y + v.z + v.w;
    }
    red[threadIdx.x] = s;
    __syncthreads();
    for (int off = 128; off > 0; off >>= 1) {
        if (threadIdx.x < off) red[threadIdx.x] += red[threadIdx.x + off];
        __syncthreads();
    }
    if (threadIdx.x == 0) g[bt] = red[0] * (1.0f / (float)POOL_N);
}

// ---------------------------------------------------------------------------
// Kernel 2: build bf16 weight copies in workspace.
//   wcb  [256 x 256]  = conv_w[o, i, 1]
//   wihb [768 x 256]  = w_ih
//   whhb [768 x 256]  = w_hh
// ---------------------------------------------------------------------------
__global__ void convert_kernel(const float* __restrict__ cw,
                               const float* __restrict__ wih,
                               const float* __restrict__ whh,
                               unsigned short* __restrict__ wcb,
                               unsigned short* __restrict__ wihb,
                               unsigned short* __restrict__ whhb) {
    int i0 = blockIdx.x * blockDim.x + threadIdx.x;
    int stride = gridDim.x * blockDim.x;
    for (int idx = i0; idx < T_DIM * T_DIM; idx += stride) {
        int o = idx >> 8, ii = idx & 255;
        wcb[idx] = f2bf(cw[o * (T_DIM * 3) + ii * 3 + 1]);
    }
    for (int idx = i0; idx < G3 * T_DIM; idx += stride) wihb[idx] = f2bf(wih[idx]);
    for (int idx = i0; idx < G3 * T_DIM; idx += stride) whhb[idx] = f2bf(whh[idx]);
}

// ---------------------------------------------------------------------------
// Kernel 3: mask GEMM + gi GEMM + 256-step GRU recurrence.
// Single workgroup, 512 threads = exactly 16 wave32s (4 waves / SIMD, one
// resident WG). Wave w owns N-tiles 3w..3w+2 of the (16 x 768) gh GEMM;
// w_hh bf16 fragments (3 x 8 x 8 = 192 VGPRs) stay register-resident for the
// whole recurrence. h lives in fusion-thread registers (exact fp32); a
// packed-bf16 copy in A-fragment layout lives in LDS so the WMMA waves load
// fragments as two ds_load_b128 each (no per-step conversion VALU).
// Steady-state per wave per step: 16 ds_load_b128 + 24 v_wmma + 24 ds_store
// + small fusion tail + 2 barriers.
// ---------------------------------------------------------------------------
__global__ void __launch_bounds__(512, 1)
gru_kernel(const float* __restrict__ g,
           const unsigned int* __restrict__ wc_u,
           const unsigned int* __restrict__ wih_u,
           const unsigned int* __restrict__ whh_u,
           const float* __restrict__ conv_b,
           const float* __restrict__ b_ih,
           const float* __restrict__ b_hh,
           float* __restrict__ out) {
    __shared__ float        sh_g[16 * G3];    // mask / gi / gh staging (48 KB)
    __shared__ unsigned int sh_hb[16 * 128];  // h as packed bf16 pairs (8 KB)

    const int tid  = threadIdx.x;
    const int lane = tid & 31;
    const int w    = tid >> 5;           // wave id 0..15 (all 16 used)
    const int nc   = lane & 15;
    const int half = lane >> 4;

    // ---- Phase 1: mask = g @ Wc^T + conv_b  (16 waves, one N-tile each)
    // mask staged in low region of sh_g with row stride 256.
    {
        int n0 = w * 16;
        v8f acc = {};
#pragma unroll
        for (int kc = 0; kc < 8; ++kc) {
            v16bf a = make_a_frag_f32(g, T_DIM, lane, kc * 32);
            v16bf b = load_b_frag(wc_u, lane, n0, kc * 32);
            acc = wmma_bf16(a, b, acc);
        }
        float cb = conv_b[n0 + nc];
#pragma unroll
        for (int r = 0; r < 8; ++r)
            sh_g[(r + 8 * half) * T_DIM + n0 + nc] = acc[r] + cb;
    }
    __syncthreads();

    // ---- Phase 2: gi = mask @ w_ih^T + b_ih (accumulate in regs, then
    // barrier, then overwrite sh_g with gi at row stride 768).
    v8f acc[3] = {v8f{}, v8f{}, v8f{}};
#pragma unroll
    for (int kc = 0; kc < 8; ++kc) {
        v16bf a = make_a_frag_f32(sh_g, T_DIM, lane, kc * 32);
#pragma unroll
        for (int t = 0; t < 3; ++t) {
            v16bf b = load_b_frag(wih_u, lane, 16 * (3 * w + t), kc * 32);
            acc[t] = wmma_bf16(a, b, acc[t]);
        }
    }
    __syncthreads();   // all mask reads done before gi overwrites sh_g
#pragma unroll
    for (int t = 0; t < 3; ++t) {
        int n0 = 16 * (3 * w + t);
        float bi = b_ih[n0 + nc];
#pragma unroll
        for (int r = 0; r < 8; ++r)
            sh_g[(r + 8 * half) * G3 + n0 + nc] = acc[t][r] + bi;
    }
    __syncthreads();

    // ---- Cache gi / b_hh / h in registers for the fused gate update.
    // Thread t handles hidden unit j = t&255 for batches b in [8*(t>>8), +8).
    const int j  = tid & 255;
    const int bg = tid >> 8;             // 0 or 1
    float gir[8], giz[8], gin[8], hreg[8];
#pragma unroll
    for (int bi = 0; bi < 8; ++bi) {
        int b = bg * 8 + bi;
        gir[bi]  = sh_g[b * G3 + j];
        giz[bi]  = sh_g[b * G3 + T_DIM + j];
        gin[bi]  = sh_g[b * G3 + 2 * T_DIM + j];
        hreg[bi] = 0.0f;                 // h0 = 0
    }
    const float bhr = b_hh[j];
    const float bhz = b_hh[T_DIM + j];
    const float bhn = b_hh[2 * T_DIM + j];

    // zero packed-bf16 h
    for (int idx = tid; idx < 16 * 128; idx += 512) sh_hb[idx] = 0u;

    // ---- Preload persistent w_hh bf16 fragments into VGPRs.
    v16bf breg[3][8];
#pragma unroll
    for (int t = 0; t < 3; ++t)
#pragma unroll
        for (int kc = 0; kc < 8; ++kc)
            breg[t][kc] = load_b_frag(whh_u, lane, 16 * (3 * w + t), kc * 32);
    __syncthreads();

    unsigned short* hb16 = (unsigned short*)sh_hb;

    // ---- 256 recurrence steps: gh = h @ w_hh^T (WMMA), barrier, fused gates.
    for (int s = 0; s < 256; ++s) {
        v8f acc0 = {}, acc1 = {}, acc2 = {};
#pragma unroll
        for (int kc = 0; kc < 8; ++kc) {
            v16bf a = load_a_packed(sh_hb, lane, kc * 32);
            acc0 = wmma_bf16(a, breg[0][kc], acc0);
            acc1 = wmma_bf16(a, breg[1][kc], acc1);
            acc2 = wmma_bf16(a, breg[2][kc], acc2);
        }
        {
            int n0 = 16 * (3 * w + 0) + nc;
            int n1 = 16 * (3 * w + 1) + nc;
            int n2 = 16 * (3 * w + 2) + nc;
#pragma unroll
            for (int r = 0; r < 8; ++r) {
                int row = (r + 8 * half) * G3;
                sh_g[row + n0] = acc0[r];
                sh_g[row + n1] = acc1[r];
                sh_g[row + n2] = acc2[r];
            }
        }
        __syncthreads();

#pragma unroll
        for (int bi = 0; bi < 8; ++bi) {
            int b = bg * 8 + bi;
            float ghr = sh_g[b * G3 + j] + bhr;
            float ghz = sh_g[b * G3 + T_DIM + j] + bhz;
            float ghn = sh_g[b * G3 + 2 * T_DIM + j] + bhn;
            float r = sigm(gir[bi] + ghr);
            float z = sigm(giz[bi] + ghz);
            float n = tanhf(gin[bi] + r * ghn);
            float hn = (1.0f - z) * n + z * hreg[bi];
            hreg[bi] = hn;                       // exact fp32 h
            hb16[b * T_DIM + j] = f2bf(hn);      // bf16 copy for WMMA A frags
            out[(size_t)b * (256 * T_DIM) + (size_t)s * T_DIM + j] = hn;
        }
        __syncthreads();
    }
}

// ---------------------------------------------------------------------------
// Launch
// ---------------------------------------------------------------------------
extern "C" void kernel_launch(void* const* d_in, const int* in_sizes, int n_in,
                              void* d_out, int out_size, void* d_ws, size_t ws_size,
                              hipStream_t stream) {
    const float* x      = (const float*)d_in[0];
    const float* conv_w = (const float*)d_in[1];
    const float* conv_b = (const float*)d_in[2];
    const float* w_ih   = (const float*)d_in[3];
    const float* w_hh   = (const float*)d_in[4];
    const float* b_ih   = (const float*)d_in[5];
    const float* b_hh   = (const float*)d_in[6];
    float* out = (float*)d_out;

    // workspace layout (bytes):
    //   [0, 16K)          g        4096 f32
    //   [16K, 144K)       wcb      256*256 bf16
    //   [144K, 528K)      wihb     768*256 bf16
    //   [528K, 912K)      whhb     768*256 bf16
    char* ws = (char*)d_ws;
    float*          gbuf = (float*)ws;
    unsigned short* wcb  = (unsigned short*)(ws + 16384);
    unsigned short* wihb = (unsigned short*)(ws + 16384 + 131072);
    unsigned short* whhb = (unsigned short*)(ws + 16384 + 131072 + 393216);

    pool_kernel<<<BT, 256, 0, stream>>>(x, gbuf);
    convert_kernel<<<256, 256, 0, stream>>>(conv_w, w_ih, w_hh, wcb, wihb, whhb);
    gru_kernel<<<1, 512, 0, stream>>>(gbuf,
                                      (const unsigned int*)wcb,
                                      (const unsigned int*)wihb,
                                      (const unsigned int*)whhb,
                                      conv_b, b_ih, b_hh, out);
}
